// Model_34093450396347
// MI455X (gfx1250) — compile-verified, matrix operand order checked by reading, not run on hardware
//
#include <hip/hip_runtime.h>

// Problem dims (fixed by the reference)
#define BSZ     256   // batch
#define NNODES  128   // nodes per tree
#define KCH     8     // max children
#define EDIM    256   // embedding dim
#define HDIM    256   // hidden dim
#define CD      512   // combined dim = E + H
#define OUTN    1024  // 3H (iou) + H (f)

typedef __attribute__((ext_vector_type(16))) __bf16 v16bf;
typedef __attribute__((ext_vector_type(8)))  __bf16 v8bf;
typedef __attribute__((ext_vector_type(8)))  float  v8f;

#define WMMA_BF16(A, Bm, C) \
    __builtin_amdgcn_wmma_f32_16x16x32_bf16(false, (A), false, (Bm), (short)0, (C), false, false)

// round-to-nearest-even float -> bf16 (bit pattern as ushort)
__device__ __forceinline__ unsigned short f2bf(float f) {
    union { float f; unsigned u; } x; x.f = f;
    unsigned u = x.u;
    unsigned r = (u + 0x7FFFu + ((u >> 16) & 1u)) >> 16;
    return (unsigned short)r;
}

__device__ __forceinline__ float sigmoidf(float x) {
    return 1.0f / (1.0f + __expf(-x));
}

// Load one 16x32 bf16 A/B fragment for the wave32 layout:
// lane l holds row (l&15), K = 8*(l>>4) + {0..7} and {16..23}.
// p is the lane-specific base; 16B aligned. Works for global or LDS pointers.
template <typename P>
__device__ __forceinline__ v16bf load_frag(P p) {
    v8bf lo = *reinterpret_cast<const v8bf*>(p);
    v8bf hi = *reinterpret_cast<const v8bf*>(p + 16);
    v16bf r;
#pragma unroll
    for (int i = 0; i < 8; ++i) { r[i] = lo[i]; r[i + 8] = hi[i]; }
    return r;
}

// ---- Kernel 1: once per launch, build Wt[n][k] = bf16 of [W_iou | W_f]^T ----
// Wt is N-major (1024 x 512) so B fragments are contiguous 16B runs.
__global__ void convert_weights(const float* __restrict__ W_iou,
                                const float* __restrict__ W_f,
                                unsigned short* __restrict__ Wt) {
    int t = blockIdx.x * blockDim.x + threadIdx.x;
    if (t >= OUTN * CD) return;
    int n = t / CD, k = t - n * CD;
    float v = (n < 3 * HDIM) ? W_iou[k * (3 * HDIM) + n]
                             : W_f[k * HDIM + (n - 3 * HDIM)];
    Wt[n * CD + k] = f2bf(v);
}

// ---- Kernel 2 (per node): fully fused step ----
// Grid: 16 blocks x 256 threads (8 waves).
// Block (mt2, nhalf): batches [mt2*32, +32), hidden cols [nhalf*128, +128).
// Phase 1: build 32x512 bf16 A-tile (embedding | sum of children h) and
//          32x128 fp32 c_sum tile in LDS (gather from prior steps' state).
// Phase 2: each wave = one 32x16 output tile: 2 M-tiles x 4 gates = 8 fp32
//          accumulators. Per K-step: 2 A frags (LDS) + 4 B frags (global)
//          feed 8 WMMAs -> 1.0 global b128 loads per WMMA. Epilogue applies
//          gates and writes h/c state (and d_out at node 0).
__global__ void __launch_bounds__(256) tree_lstm_step(
        const int*   __restrict__ ast_nodes,
        const int*   __restrict__ children,
        const float* __restrict__ emb,
        const unsigned short* __restrict__ Wt,     // 1024 x 512 bf16
        const float* __restrict__ b_iou,           // 768
        const float* __restrict__ b_f,             // 256
        float* __restrict__ h_state,               // 256 x 128 x 256
        float* __restrict__ c_state,
        float* __restrict__ out,                   // 256 x 256 (written at node 0)
        int node) {
    __shared__ unsigned short sA[32 * CD];   // 32 KB: bf16 combined tile
    __shared__ float          sC[32 * 128];  // 16 KB: c_sum for this block's cols

    const int tid   = threadIdx.x;
    const int mt2   = blockIdx.x >> 1;       // 0..7 over batch (32 rows each)
    const int nhalf = blockIdx.x & 1;        // which 128-col half of H

    // ---- Phase 1: prepare tiles in LDS ----
    // A-tile: 32 rows x 512 cols = 16384 entries, 64 per thread.
    for (int e = tid; e < 32 * CD; e += 256) {
        const int r = e >> 9;                // 0..31
        const int k = e & (CD - 1);
        const int b = mt2 * 32 + r;
        float v;
        if (k < EDIM) {
            const int tok = ast_nodes[b * NNODES + node];
            v = emb[tok * EDIM + k];
        } else {
            const int j = k - EDIM;
            const int* ch = children + (b * NNODES + node) * KCH;
            float hs = 0.f;
#pragma unroll
            for (int q = 0; q < KCH; ++q) {
                const int c = ch[q];
                if (c >= 0)   // children have index > node: already computed
                    hs += h_state[(b * NNODES + c) * HDIM + j];
            }
            v = hs;
        }
        sA[e] = f2bf(v);
    }
    // c_sum tile: 32 rows x 128 cols, 16 per thread.
    for (int e = tid; e < 32 * 128; e += 256) {
        const int r = e >> 7;
        const int j = nhalf * 128 + (e & 127);
        const int b = mt2 * 32 + r;
        const int* ch = children + (b * NNODES + node) * KCH;
        float cs = 0.f;
#pragma unroll
        for (int q = 0; q < KCH; ++q) {
            const int c = ch[q];
            if (c >= 0)
                cs += c_state[(b * NNODES + c) * HDIM + j];
        }
        sC[e] = cs;
    }
    __syncthreads();

    // ---- Phase 2: WMMA GEMM, 2 M-tiles x 4 gates per wave ----
    const int wave = tid >> 5;
    const int lane = tid & 31;
    const int nt   = nhalf * 8 + wave;       // 0..15 over H tiles
    const int lrow = lane & 15;
    const int koff = (lane >> 4) << 3;       // 0 or 8

    const unsigned short* pa0 = sA + (lrow)      * CD + koff;  // M-tile 0 (LDS)
    const unsigned short* pa1 = sA + (16 + lrow) * CD + koff;  // M-tile 1 (LDS)
    const int col = nt * 16 + lrow;
    const unsigned short* pbi = Wt + (0 * HDIM + col) * CD + koff;
    const unsigned short* pbo = Wt + (1 * HDIM + col) * CD + koff;
    const unsigned short* pbu = Wt + (2 * HDIM + col) * CD + koff;
    const unsigned short* pbf = Wt + (3 * HDIM + col) * CD + koff;

    v8f ai0 = {}; v8f ao0 = {}; v8f au0 = {}; v8f af0 = {};
    v8f ai1 = {}; v8f ao1 = {}; v8f au1 = {}; v8f af1 = {};

#pragma unroll
    for (int kk = 0; kk < CD; kk += 32) {
        v16bf a0 = load_frag(pa0 + kk);      // ds_load_b128 x2
        v16bf a1 = load_frag(pa1 + kk);      // ds_load_b128 x2
        v16bf bi = load_frag(pbi + kk);      // global_load_b128 x2 (L2 resident)
        v16bf bo = load_frag(pbo + kk);
        v16bf bu = load_frag(pbu + kk);
        v16bf bf = load_frag(pbf + kk);
        ai0 = WMMA_BF16(a0, bi, ai0);
        ai1 = WMMA_BF16(a1, bi, ai1);
        ao0 = WMMA_BF16(a0, bo, ao0);
        ao1 = WMMA_BF16(a1, bo, ao1);
        au0 = WMMA_BF16(a0, bu, au0);
        au1 = WMMA_BF16(a1, bu, au1);
        af0 = WMMA_BF16(a0, bf, af0);
        af1 = WMMA_BF16(a1, bf, af1);
    }

    // Gate epilogue. C/D layout: VGPR v holds (M = v + 8*(lane>=16), N = lane&15).
    const int j    = nt * 16 + lrow;             // global hidden index
    const int jl   = wave * 16 + lrow;           // col within block's 128-half
    const float bi_s = b_iou[j];
    const float bo_s = b_iou[HDIM + j];
    const float bu_s = b_iou[2 * HDIM + j];
    const float bf_s = b_f[j];
#pragma unroll
    for (int mh = 0; mh < 2; ++mh) {
        const v8f& qi = mh ? ai1 : ai0;
        const v8f& qo = mh ? ao1 : ao0;
        const v8f& qu = mh ? au1 : au0;
        const v8f& qf = mh ? af1 : af0;
#pragma unroll
        for (int v = 0; v < 8; ++v) {
            const int r = mh * 16 + koff + v;    // local batch row 0..31
            const int b = mt2 * 32 + r;
            float ig = sigmoidf(qi[v] + bi_s);
            float og = sigmoidf(qo[v] + bo_s);
            float ug = tanhf(qu[v] + bu_s);
            float fg = sigmoidf(qf[v] + bf_s);
            float cn = ig * ug + fg * sC[r * 128 + jl];
            float hn = og * tanhf(cn);
            h_state[(b * NNODES + node) * HDIM + j] = hn;
            c_state[(b * NNODES + node) * HDIM + j] = cn;
            if (node == 0) out[b * HDIM + j] = hn;   // final answer = h[:, 0]
        }
    }
}

extern "C" void kernel_launch(void* const* d_in, const int* in_sizes, int n_in,
                              void* d_out, int out_size, void* d_ws, size_t ws_size,
                              hipStream_t stream) {
    const int*   ast_nodes = (const int*)  d_in[0];   // 256*128
    const int*   children  = (const int*)  d_in[1];   // 256*128*8
    const float* emb_table = (const float*)d_in[2];   // 1000*256
    const float* W_iou     = (const float*)d_in[3];   // 512*768
    const float* b_iou     = (const float*)d_in[4];   // 768
    const float* W_f       = (const float*)d_in[5];   // 512*256
    const float* b_f       = (const float*)d_in[6];   // 256
    float* out = (float*)d_out;                       // 256*256

    // Workspace layout (all state written before read each call)
    char* ws = (char*)d_ws;
    size_t off = 0;
    unsigned short* Wt      = (unsigned short*)(ws + off); off += (size_t)OUTN * CD * 2;           // 1 MB
    float*          h_state = (float*)(ws + off);          off += (size_t)BSZ * NNODES * HDIM * 4; // 32 MB
    float*          c_state = (float*)(ws + off);          off += (size_t)BSZ * NNODES * HDIM * 4; // 32 MB
    (void)ws_size; (void)in_sizes; (void)n_in; (void)out_size;

    // One-time (per call) bf16 transposed weight pack: L2-resident thereafter.
    convert_weights<<<(OUTN * CD + 255) / 256, 256, 0, stream>>>(W_iou, W_f, Wt);

    // Serial scan over nodes, high index -> 0 (children precede parents).
    // One fused kernel per step: min launch count for the dependency chain.
    for (int node = NNODES - 1; node >= 0; --node) {
        tree_lstm_step<<<16, 256, 0, stream>>>(
            ast_nodes, children, emb_table, Wt, b_iou, b_f,
            h_state, c_state, out, node);
    }
}
